// Network_13632226197685
// MI455X (gfx1250) — compile-verified, matrix operand order checked by reading, not run on hardware
//
#include <hip/hip_runtime.h>
#include <hip/hip_bf16.h>
#include <math.h>

typedef __attribute__((ext_vector_type(16))) __bf16 v16bf;
typedef __attribute__((ext_vector_type(8)))  __bf16 v8bf;
typedef __attribute__((ext_vector_type(8)))  float  v8f;
typedef __attribute__((ext_vector_type(4)))  int    v4i;

#if defined(__has_builtin)
#if __has_builtin(__builtin_amdgcn_global_load_async_to_lds_b128)
#define EQ_HAS_ASYNC_LDS 1
#endif
#if __has_builtin(__builtin_amdgcn_s_wait_asynccnt)
#define EQ_HAS_WAIT_ASYNC 1
#endif
#if __has_builtin(__builtin_amdgcn_s_wait_tensorcnt)
#define EQ_HAS_WAIT_TENSOR 1
#endif
#endif

namespace {
constexpr int   NNODE = 8192;
constexpr int   NEDGE = 131072;
constexpr float MAXRR = 4.0f;
constexpr float K_CS  = 0.38268343236f;   // sin(pi/8)
constexpr float K_CX  = 0.92387953251f;   // cos(pi/8)
constexpr float K_SQ3 = 1.73205080757f;
constexpr float ISQ32 = 0.17677669529f;   // 1/sqrt(32)
constexpr float ISQ64 = 0.125f;
constexpr float ISQ96 = 0.10206207262f;
constexpr float ISQ10 = 0.31622776601f;
constexpr float ISQ100= 0.1f;
constexpr float KINV2 = 0.70710678118f;   // 1/sqrt(2)
constexpr float KINV3 = 0.57735026919f;   // 1/sqrt(3)
constexpr float SEGSC = 0.25f;            // 1/sqrt(16) neighbors
constexpr float OUTSC = 0.04419417382f;   // 1/sqrt(512)
constexpr int   LDSB_ELEMS = 16384;       // 32KB bf16 staging buffer for B
}

// gfx1250 async global->LDS copy of one 16B vector (ASYNCcnt-tracked).
#ifdef EQ_HAS_ASYNC_LDS
static __device__ __forceinline__ void eq_async_b128(const void* g, void* l) {
  __builtin_amdgcn_global_load_async_to_lds_b128(
      (__attribute__((address_space(1))) v4i*)g,
      (__attribute__((address_space(3))) v4i*)l, 0, 0);
}
#endif

// ---------------------------------------------------------------------------
// Fragment helpers (CDNA5 ISA §7.12.2):
//   A (16x32 bf16, row-major): lane L (m=L&15, h=L>>4) holds K={h*8..h*8+7}
//     then {16+h*8..16+h*8+7} -> two contiguous 16B runs per lane.
//   B is PRE-PACKED into fragment order: [ktile][ntile][lane][16] so a lane's
//     16 values are 32 contiguous bytes (two b128 loads, global or LDS).
//   C/D (16x16 f32): VGPR i holds row m0 + h*8 + i, column lane&15.
// ---------------------------------------------------------------------------
static __device__ __forceinline__ v16bf eq_frag16(const __bf16* p) {
  v8bf lo = *(const v8bf*)p;
  v8bf hi = *(const v8bf*)(p + 8);
  return __builtin_shufflevector(lo, hi, 0, 1, 2, 3, 4, 5, 6, 7,
                                 8, 9, 10, 11, 12, 13, 14, 15);
}

static __device__ __forceinline__
v16bf eq_load_a(const __bf16* __restrict__ A, int lda, int m0, int lane) {
  int h = (lane >> 4) & 1;
  int m = lane & 15;
  const __bf16* row = A + (size_t)(m0 + m) * lda;
  v8bf lo = *(const v8bf*)(row + h * 8);
  v8bf hi = *(const v8bf*)(row + 16 + h * 8);
  return __builtin_shufflevector(lo, hi, 0, 1, 2, 3, 4, 5, 6, 7,
                                 8, 9, 10, 11, 12, 13, 14, 15);
}

// Generic bf16 GEMM: C = scale*A@B (+C if accflag). B pre-packed (see above).
// Each wave computes a 16 x (16*TN) strip: one A fragment feeds TN WMMAs,
// cutting A-operand HBM traffic by TN (the dominant GEMM is A-bandwidth
// bound). Small B matrices are staged to LDS via the gfx1250 async path and
// fragments then read with ds_load_b128.
template <int TN>
__global__ void eq_gemm(const __bf16* __restrict__ A, const __bf16* __restrict__ B,
                        float* __restrict__ C, int Mr, int Nc, int Kd,
                        int lda, int ldc, float scale, int accflag) {
  __shared__ __attribute__((aligned(16))) __bf16 Bs[LDSB_ELEMS];
  const int ntiles = Nc >> 4;
  const int bElems = Kd * Nc;               // packed B element count
  const bool useLds = bElems <= LDSB_ELEMS; // uniform across block

  if (useLds) {
    int nvec = bElems >> 3;                 // 16B vectors
    for (int i = threadIdx.x; i < nvec; i += blockDim.x) {
#ifdef EQ_HAS_ASYNC_LDS
      eq_async_b128(((const uint4*)B) + i, ((uint4*)Bs) + i);
#else
      ((uint4*)Bs)[i] = ((const uint4*)B)[i];
#endif
    }
#ifdef EQ_HAS_WAIT_ASYNC
    __builtin_amdgcn_s_wait_asynccnt(0);
#endif
    __syncthreads();
  }

  int wave = blockIdx.x * (blockDim.x >> 5) + (threadIdx.x >> 5);
  int lane = threadIdx.x & 31;
  int ngroups = ntiles / TN;
  int total = (Mr >> 4) * ngroups;
  if (wave >= total) return;
  int m0     = (wave / ngroups) << 4;
  int ntBase = (wave % ngroups) * TN;

  v8f acc[TN];
  v8f z = {0.f, 0.f, 0.f, 0.f, 0.f, 0.f, 0.f, 0.f};
#pragma unroll
  for (int t = 0; t < TN; ++t) acc[t] = z;

  if (useLds) {
    for (int k0 = 0; k0 < Kd; k0 += 32) {
      v16bf a = eq_load_a(A + k0, lda, m0, lane);
      size_t kbase = (size_t)(k0 >> 5) * ntiles;
#pragma unroll
      for (int t = 0; t < TN; ++t) {
        v16bf b = eq_frag16(Bs + ((kbase + ntBase + t) * 32 + lane) * 16);
        acc[t] = __builtin_amdgcn_wmma_f32_16x16x32_bf16(false, a, false, b,
                                                         (short)0, acc[t], false, false);
      }
    }
  } else {
    for (int k0 = 0; k0 < Kd; k0 += 32) {
      if (k0 + 32 < Kd)   // WGP-scope prefetch of next A k-block
        __builtin_prefetch(A + (size_t)(m0 + (lane & 15)) * lda + k0 + 32, 0, 3);
      v16bf a = eq_load_a(A + k0, lda, m0, lane);
      size_t kbase = (size_t)(k0 >> 5) * ntiles;
#pragma unroll
      for (int t = 0; t < TN; ++t) {
        v16bf b = eq_frag16(B + ((kbase + ntBase + t) * 32 + lane) * 16);
        acc[t] = __builtin_amdgcn_wmma_f32_16x16x32_bf16(false, a, false, b,
                                                         (short)0, acc[t], false, false);
      }
    }
  }

  int h = (lane >> 4) & 1;
  int n = lane & 15;
#pragma unroll
  for (int t = 0; t < TN; ++t) {
    float* cp = C + (size_t)(m0 + h * 8) * ldc + ((ntBase + t) << 4) + n;
    if (accflag) {
#pragma unroll
      for (int i = 0; i < 8; ++i) cp[(size_t)i * ldc] += acc[t][i] * scale;
    } else {
#pragma unroll
      for (int i = 0; i < 8; ++i) cp[(size_t)i * ldc] = acc[t][i] * scale;
    }
  }
}

// Radial hidden layer: D(bf16) = silu(scale * A@B); B always LDS-resident.
__global__ void eq_gemm_silu(const __bf16* __restrict__ A, const __bf16* __restrict__ B,
                             __bf16* __restrict__ D, int Mr, int Nc, int Kd,
                             int lda, int ldd, float scale) {
  __shared__ __attribute__((aligned(16))) __bf16 Bs[LDSB_ELEMS];
  const int ntiles = Nc >> 4;
  const int bElems = Kd * Nc;
  {
    int nvec = bElems >> 3;
    for (int i = threadIdx.x; i < nvec; i += blockDim.x) {
#ifdef EQ_HAS_ASYNC_LDS
      eq_async_b128(((const uint4*)B) + i, ((uint4*)Bs) + i);
#else
      ((uint4*)Bs)[i] = ((const uint4*)B)[i];
#endif
    }
#ifdef EQ_HAS_WAIT_ASYNC
    __builtin_amdgcn_s_wait_asynccnt(0);
#endif
    __syncthreads();
  }

  int wave = blockIdx.x * (blockDim.x >> 5) + (threadIdx.x >> 5);
  int lane = threadIdx.x & 31;
  int total = (Mr >> 4) * ntiles;
  if (wave >= total) return;
  int m0 = (wave / ntiles) << 4;
  int nt = wave % ntiles;

  v8f acc = {0.f, 0.f, 0.f, 0.f, 0.f, 0.f, 0.f, 0.f};
  for (int k0 = 0; k0 < Kd; k0 += 32) {
    v16bf a = eq_load_a(A + k0, lda, m0, lane);
    v16bf b = eq_frag16(Bs + (((size_t)(k0 >> 5) * ntiles + nt) * 32 + lane) * 16);
    acc = __builtin_amdgcn_wmma_f32_16x16x32_bf16(false, a, false, b,
                                                  (short)0, acc, false, false);
  }

  int h = (lane >> 4) & 1;
  int n = lane & 15;
  __bf16* dp = D + (size_t)(m0 + h * 8) * ldd + (nt << 4) + n;
#pragma unroll
  for (int i = 0; i < 8; ++i) {
    float v = acc[i] * scale;
    v = v / (1.0f + expf(-v));           // silu
    dp[(size_t)i * ldd] = (__bf16)v;
  }
}

// ---------------------------------------------------------------------------
// Elementwise / irregular kernels
// ---------------------------------------------------------------------------
__global__ void eq_zero_f32(float* __restrict__ p, long long n) {
  long long i = (long long)blockIdx.x * blockDim.x + threadIdx.x;
  if (i < n) p[i] = 0.0f;
}

// Pack f32 matrix into zero-padded row-major bf16 (A operands).
__global__ void eq_pack(const float* __restrict__ src, __bf16* __restrict__ dst,
                        int srows, int scols, int drows, int dcols) {
  int i = blockIdx.x * blockDim.x + threadIdx.x;
  if (i >= drows * dcols) return;
  int r = i / dcols, c = i % dcols;
  float v = (r < srows && c < scols) ? src[(size_t)r * scols + c] : 0.0f;
  dst[i] = (__bf16)v;
}

// Pack f32 (srows x scols) weight into fragment-blocked bf16 B of logical size
// Kp x Np: dst[((kt*ntiles + nt)*32 + lane)*16 + j] = W[kt*32+kk, nt*16+(lane&15)]
// with kk = (j<8 ? h*8+j : 16+h*8+j-8), h = lane>>4. Zero-padded outside.
__global__ void eq_packB(const float* __restrict__ src, __bf16* __restrict__ dst,
                         int srows, int scols, int Kp, int Np) {
  int i = blockIdx.x * blockDim.x + threadIdx.x;
  if (i >= Kp * Np) return;
  int j    = i & 15;
  int lane = (i >> 4) & 31;
  int t    = i >> 9;
  int ntiles = Np >> 4;
  int nt = t % ntiles;
  int kt = t / ntiles;
  int h  = lane >> 4;
  int kk = (j < 8) ? (h * 8 + j) : (16 + h * 8 + (j - 8));
  int r = kt * 32 + kk;
  int c = nt * 16 + (lane & 15);
  float v = (r < srows && c < scols) ? src[(size_t)r * scols + c] : 0.0f;
  dst[i] = (__bf16)v;
}

// Split x(N,256) into s_o,s_e (N,32) and plane-major v_o,v_e ([3][N][32]).
__global__ void eq_state_init(const float* __restrict__ x,
                              float* __restrict__ so, float* __restrict__ se,
                              float* __restrict__ vo, float* __restrict__ ve) {
  int i = blockIdx.x * blockDim.x + threadIdx.x;
  if (i >= NNODE * 32) return;
  int n = i >> 5, u = i & 31;
  const float* row = x + (size_t)n * 256;
  so[i] = row[u];
  se[i] = row[32 + u];
  const size_t PS = (size_t)NNODE * 32;
#pragma unroll
  for (int p = 0; p < 3; ++p) {
    vo[p * PS + i] = row[64 + u * 3 + p];
    ve[p * PS + i] = row[160 + u * 3 + p];
  }
}

// Per-edge geometry: cutoff y0, l=1 basis y1, radial embedding (bf16, padded).
__global__ void eq_geom(const float* __restrict__ ev, float* __restrict__ y0,
                        float* __restrict__ y1, __bf16* __restrict__ emb) {
  int e = blockIdx.x * blockDim.x + threadIdx.x;
  if (e >= NEDGE) return;
  float ex = ev[e * 3], ey = ev[e * 3 + 1], ez = ev[e * 3 + 2];
  float r = sqrtf(ex * ex + ey * ey + ez * ez);
  float inv = 1.0f / (r + 1e-12f);
  float u = 2.0f * (r * (1.0f / MAXRR) - 1.0f);
  float cut = (u > 0.0f) ? 0.0f
            : ((u < -1.0f) ? 1.0f : 0.5f * (1.0f - cosf(3.14159265358979f * u)));
  y0[e] = cut;
  float s = K_SQ3 * cut * inv;
  y1[e * 3] = s * ex; y1[e * 3 + 1] = s * ey; y1[e * 3 + 2] = s * ez;
  const float step = MAXRR / 9.0f;
#pragma unroll
  for (int j = 0; j < 32; ++j) {
    float v = 0.0f;
    if (j < 10) {
      float d = (r - step * j) / step;
      v = expf(-d * d) * (3.16227766017f / 1.12f);
    }
    emb[(size_t)e * 32 + j] = (__bf16)v;
  }
#ifdef EQ_HAS_WAIT_TENSOR
  __builtin_amdgcn_s_wait_tensorcnt(0);
#endif
}

// Tensor-product messages + segment-sum scatter, one thread per (edge,channel).
__global__ void eq_msg(const float* __restrict__ a_so, const float* __restrict__ a_se,
                       const float* __restrict__ a_vo, const float* __restrict__ a_ve,
                       const float* __restrict__ W,    const float* __restrict__ y0,
                       const float* __restrict__ y1,
                       const int* __restrict__ esrc,   const int* __restrict__ edst,
                       float* __restrict__ n0o, float* __restrict__ n0e,
                       float* __restrict__ n1o, float* __restrict__ n1e) {
  int idx = blockIdx.x * blockDim.x + threadIdx.x;
  if (idx >= NEDGE * 32) return;
  int e = idx >> 5, u = idx & 31;
  int s = esrc[e], d = edst[e];
  const size_t PS = (size_t)NNODE * 32;
  float yy0 = y0[e];
  float yc[3] = {y1[e * 3], y1[e * 3 + 1], y1[e * 3 + 2]};
  float eso = a_so[(size_t)s * 32 + u];
  float ese = a_se[(size_t)s * 32 + u];
  float vo[3], ve[3];
#pragma unroll
  for (int p = 0; p < 3; ++p) {
    vo[p] = a_vo[p * PS + (size_t)s * 32 + u];
    ve[p] = a_ve[p * PS + (size_t)s * 32 + u];
  }
  const float* wr = W + (size_t)e * 320;
  float w[10];
#pragma unroll
  for (int k = 0; k < 10; ++k) w[k] = wr[k * 32 + u];

  float dot_ve = (ve[0]*yc[0] + ve[1]*yc[1] + ve[2]*yc[2]) * KINV3;
  float dot_vo = (vo[0]*yc[0] + vo[1]*yc[1] + vo[2]*yc[2]) * KINV3;
  float crve[3] = {(ve[1]*yc[2] - ve[2]*yc[1]) * KINV2,
                   (ve[2]*yc[0] - ve[0]*yc[2]) * KINV2,
                   (ve[0]*yc[1] - ve[1]*yc[0]) * KINV2};
  float crvo[3] = {(vo[1]*yc[2] - vo[2]*yc[1]) * KINV2,
                   (vo[2]*yc[0] - vo[0]*yc[2]) * KINV2,
                   (vo[0]*yc[1] - vo[1]*yc[0]) * KINV2};
  const float c2 = KINV2 * SEGSC;
  const float c3 = KINV3 * SEGSC;

  atomicAdd(&n0o[(size_t)d * 64 + u],      w[0] * eso * yy0 * c2);
  atomicAdd(&n0o[(size_t)d * 64 + 32 + u], w[1] * dot_ve   * c2);
  atomicAdd(&n0e[(size_t)d * 64 + u],      w[2] * ese * yy0 * c2);
  atomicAdd(&n0e[(size_t)d * 64 + 32 + u], w[3] * dot_vo   * c2);

  const size_t VPS = (size_t)NNODE * 96;
#pragma unroll
  for (int p = 0; p < 3; ++p) {
    float* po = n1o + p * VPS + (size_t)d * 96;
    float* pe = n1e + p * VPS + (size_t)d * 96;
    atomicAdd(&po[u],      w[4] * ese * yc[p] * c3);
    atomicAdd(&po[32 + u], w[5] * vo[p] * yy0 * c3);
    atomicAdd(&po[64 + u], w[6] * crve[p]     * c3);
    atomicAdd(&pe[u],      w[7] * eso * yc[p] * c3);
    atomicAdd(&pe[32 + u], w[8] * crvo[p]     * c3);
    atomicAdd(&pe[64 + u], w[9] * ve[p] * yy0 * c3);
  }
}

// Gates + activations -> next layer state.
__global__ void eq_nonlin(const float* __restrict__ o0o, const float* __restrict__ o0e,
                          const float* __restrict__ o1o, const float* __restrict__ o1e,
                          float* __restrict__ so, float* __restrict__ se,
                          float* __restrict__ vo, float* __restrict__ ve) {
  int i = blockIdx.x * blockDim.x + threadIdx.x;
  if (i >= NNODE * 32) return;
  int n = i >> 5, u = i & 31;
  so[i] = tanhf(o0o[(size_t)n * 32 + u]);
  float b = o0e[(size_t)n * 96 + u];
  se[i] = b / (1.0f + expf(-b));
  float g1 = 1.0f / (1.0f + expf(-o0e[(size_t)n * 96 + 32 + u]));
  float g2 = 1.0f / (1.0f + expf(-o0e[(size_t)n * 96 + 64 + u]));
  const size_t PS = (size_t)NNODE * 32;
#pragma unroll
  for (int p = 0; p < 3; ++p) {
    vo[p * PS + i] = o1o[p * PS + i] * g1;
    ve[p * PS + i] = o1e[p * PS + i] * g2;
  }
}

// Final edge message (scalar-even channel only).
__global__ void eq_fmsg(const float* __restrict__ a_se, const float* __restrict__ a_vo,
                        const float* __restrict__ W,    const float* __restrict__ y0,
                        const float* __restrict__ y1,
                        const int* __restrict__ esrc,   const int* __restrict__ edst,
                        float* __restrict__ n0e) {
  int idx = blockIdx.x * blockDim.x + threadIdx.x;
  if (idx >= NEDGE * 32) return;
  int e = idx >> 5, u = idx & 31;
  int s = esrc[e], d = edst[e];
  const size_t PS = (size_t)NNODE * 32;
  float ase = a_se[(size_t)s * 32 + u];
  float vox = a_vo[(size_t)s * 32 + u];
  float voy = a_vo[PS + (size_t)s * 32 + u];
  float voz = a_vo[2 * PS + (size_t)s * 32 + u];
  float y1x = y1[e * 3], y1y = y1[e * 3 + 1], y1z = y1[e * 3 + 2];
  float dot_vo = (vox * y1x + voy * y1y + voz * y1z) * KINV3;
  float w0 = W[(size_t)e * 64 + u];
  float w1 = W[(size_t)e * 64 + 32 + u];
  const float c2 = KINV2 * SEGSC;
  atomicAdd(&n0e[(size_t)d * 64 + u],      w0 * ase * y0[e] * c2);
  atomicAdd(&n0e[(size_t)d * 64 + 32 + u], w1 * dot_vo * c2);
}

// Per-node readout + graph segment sum.
__global__ void eq_readout(const float* __restrict__ se, const float* __restrict__ n0e,
                           const float* __restrict__ fsc, const float* __restrict__ fl2,
                           const int* __restrict__ batch, float* __restrict__ out) {
  int n = blockIdx.x * blockDim.x + threadIdx.x;
  if (n >= NNODE) return;
  float a1 = 0.0f, a2 = 0.0f;
  for (int u = 0; u < 32; ++u) a1 += se[(size_t)n * 32 + u] * fsc[u];
  for (int c = 0; c < 64; ++c) a2 += n0e[(size_t)n * 64 + c] * fl2[c];
  float v = K_CS * ISQ32 * a1 + K_CX * ISQ64 * a2;
  atomicAdd(&out[batch[n]], v * OUTSC);
}

// ---------------------------------------------------------------------------
// Host orchestration
// ---------------------------------------------------------------------------
extern "C" void kernel_launch(void* const* d_in, const int* in_sizes, int n_in,
                              void* d_out, int out_size, void* d_ws, size_t ws_size,
                              hipStream_t stream) {
  (void)in_sizes; (void)n_in; (void)out_size; (void)ws_size;
  const float* x        = (const float*)d_in[0];
  const float* edge_vec = (const float*)d_in[1];
  const float* lin1_so  = (const float*)d_in[2];
  const float* lin1_se  = (const float*)d_in[3];
  const float* lin1_vo  = (const float*)d_in[4];
  const float* lin1_ve  = (const float*)d_in[5];
  const float* sc_so    = (const float*)d_in[6];
  const float* sc_se    = (const float*)d_in[7];
  const float* sc_vo    = (const float*)d_in[8];
  const float* sc_ve    = (const float*)d_in[9];
  const float* rad_w1   = (const float*)d_in[10];
  const float* rad_w2   = (const float*)d_in[11];
  const float* lin2_so  = (const float*)d_in[12];
  const float* lin2_se  = (const float*)d_in[13];
  const float* lin2_vo  = (const float*)d_in[14];
  const float* lin2_ve  = (const float*)d_in[15];
  const float* flin1_se = (const float*)d_in[16];
  const float* flin1_vo = (const float*)d_in[17];
  const float* fsc_se   = (const float*)d_in[18];
  const float* frad_w1  = (const float*)d_in[19];
  const float* frad_w2  = (const float*)d_in[20];
  const float* flin2    = (const float*)d_in[21];
  const int*   esrc     = (const int*)d_in[22];
  const int*   edst     = (const int*)d_in[23];
  const int*   batch    = (const int*)d_in[24];
  float* out = (float*)d_out;

  char* base = (char*)d_ws;
  size_t off = 0;
  auto alloc = [&](size_t bytes) -> void* {
    off = (off + 255) & ~(size_t)255;
    void* p = base + off; off += bytes; return p;
  };
  const size_t PS  = (size_t)NNODE * 32;   // scalar plane (f32 elems)
  const size_t VS  = 3 * PS;               // vector planes

  // f32 state (ping-pong)
  float* sa_so = (float*)alloc(PS * 4);  float* sa_se = (float*)alloc(PS * 4);
  float* sa_vo = (float*)alloc(VS * 4);  float* sa_ve = (float*)alloc(VS * 4);
  float* sb_so = (float*)alloc(PS * 4);  float* sb_se = (float*)alloc(PS * 4);
  float* sb_vo = (float*)alloc(VS * 4);  float* sb_ve = (float*)alloc(VS * 4);
  // lin1 outputs
  float* a_so = (float*)alloc(PS * 4);   float* a_se = (float*)alloc(PS * 4);
  float* a_vo = (float*)alloc(VS * 4);   float* a_ve = (float*)alloc(VS * 4);
  // output linear results
  float* o0o = (float*)alloc(PS * 4);
  float* o0e = (float*)alloc((size_t)NNODE * 96 * 4);
  float* o1o = (float*)alloc(VS * 4);
  float* o1e = (float*)alloc(VS * 4);
  // geometry
  float* y0 = (float*)alloc((size_t)NEDGE * 4);
  float* y1 = (float*)alloc((size_t)NEDGE * 3 * 4);
  // message accumulators (contiguous so one zero pass clears all)
  float* n0o = (float*)alloc((size_t)NNODE * 64 * 4);
  float* n0e = (float*)alloc((size_t)NNODE * 64 * 4);
  float* n1o = (float*)alloc((size_t)NNODE * 96 * 3 * 4);
  float* n1e = (float*)alloc((size_t)NNODE * 96 * 3 * 4);
  const long long accTotal = (long long)NNODE * 64 * 2 + (long long)NNODE * 96 * 3 * 2;
  // radial tensors
  float* Wrad = (float*)alloc((size_t)NEDGE * 320 * 4);
  // bf16 operands (A side, row-major)
  __bf16* bf_so  = (__bf16*)alloc(PS * 2);  __bf16* bf_se  = (__bf16*)alloc(PS * 2);
  __bf16* bf_vo  = (__bf16*)alloc(VS * 2);  __bf16* bf_ve  = (__bf16*)alloc(VS * 2);
  __bf16* bf_n0o = (__bf16*)alloc((size_t)NNODE * 64 * 2);
  __bf16* bf_n0e = (__bf16*)alloc((size_t)NNODE * 64 * 2);
  __bf16* bf_n1o = (__bf16*)alloc((size_t)NNODE * 96 * 3 * 2);
  __bf16* bf_n1e = (__bf16*)alloc((size_t)NNODE * 96 * 3 * 2);
  __bf16* emb_bf = (__bf16*)alloc((size_t)NEDGE * 32 * 2);
  __bf16* H_bf   = (__bf16*)alloc((size_t)NEDGE * 128 * 2);
  // bf16 weights (fragment-blocked, padded)
  __bf16* wb_l1so = (__bf16*)alloc(32 * 32 * 2);
  __bf16* wb_l1se = (__bf16*)alloc(32 * 32 * 2);
  __bf16* wb_l1vo = (__bf16*)alloc(32 * 32 * 2);
  __bf16* wb_l1ve = (__bf16*)alloc(32 * 32 * 2);
  __bf16* wb_wsso = (__bf16*)alloc(32 * 32 * 2);
  __bf16* wb_wsse = (__bf16*)alloc(32 * 96 * 2);
  __bf16* wb_wsvo = (__bf16*)alloc(32 * 32 * 2);
  __bf16* wb_wsve = (__bf16*)alloc(32 * 32 * 2);
  __bf16* wb_rw1  = (__bf16*)alloc(32 * 112 * 2);
  __bf16* wb_rw2  = (__bf16*)alloc(128 * 320 * 2);
  __bf16* wb_w2so = (__bf16*)alloc(64 * 32 * 2);
  __bf16* wb_w2se = (__bf16*)alloc(64 * 96 * 2);
  __bf16* wb_w2vo = (__bf16*)alloc(96 * 32 * 2);
  __bf16* wb_w2ve = (__bf16*)alloc(96 * 32 * 2);
  __bf16* wb_f1se = (__bf16*)alloc(32 * 32 * 2);
  __bf16* wb_f1vo = (__bf16*)alloc(32 * 32 * 2);
  __bf16* wb_frw1 = (__bf16*)alloc(32 * 112 * 2);
  __bf16* wb_frw2 = (__bf16*)alloc(128 * 64 * 2);

  auto cdiv = [](long long a, long long b) { return (unsigned)((a + b - 1) / b); };
  auto zero = [&](float* p, long long n) {
    eq_zero_f32<<<cdiv(n, 256), 256, 0, stream>>>(p, n);
  };
  auto pack = [&](const float* s, __bf16* d, int sr, int sc, int dr, int dc) {
    eq_pack<<<cdiv((long long)dr * dc, 256), 256, 0, stream>>>(s, d, sr, sc, dr, dc);
  };
  auto packB = [&](const float* s, __bf16* d, int sr, int sc, int Kp, int Np) {
    eq_packB<<<cdiv((long long)Kp * Np, 256), 256, 0, stream>>>(s, d, sr, sc, Kp, Np);
  };
  auto gemm = [&](const __bf16* A, const __bf16* B, float* C, int Mr, int Nc, int Kd,
                  int lda, int ldc, float scale, int acc) {
    int ntiles = Nc / 16;
    int TN = (ntiles % 4 == 0) ? 4 : ((ntiles % 2 == 0) ? 2 : 1);
    long long groups = (long long)(Mr / 16) * (ntiles / TN);
    unsigned blocks = cdiv(groups, 8);
    if (TN == 4)
      eq_gemm<4><<<blocks, 256, 0, stream>>>(A, B, C, Mr, Nc, Kd, lda, ldc, scale, acc);
    else if (TN == 2)
      eq_gemm<2><<<blocks, 256, 0, stream>>>(A, B, C, Mr, Nc, Kd, lda, ldc, scale, acc);
    else
      eq_gemm<1><<<blocks, 256, 0, stream>>>(A, B, C, Mr, Nc, Kd, lda, ldc, scale, acc);
  };
  auto gemm_silu = [&](const __bf16* A, const __bf16* B, __bf16* D, int Mr, int Nc, int Kd,
                       int lda, int ldd, float scale) {
    long long tiles = (long long)(Mr / 16) * (Nc / 16);
    eq_gemm_silu<<<cdiv(tiles, 8), 256, 0, stream>>>(A, B, D, Mr, Nc, Kd, lda, ldd, scale);
  };

  // --- setup ---
  eq_state_init<<<cdiv((long long)NNODE * 32, 256), 256, 0, stream>>>(x, sa_so, sa_se, sa_vo, sa_ve);
  eq_geom<<<cdiv(NEDGE, 256), 256, 0, stream>>>(edge_vec, y0, y1, emb_bf);
  zero((float*)H_bf, (long long)NEDGE * 128 / 2);   // also clears K-pad cols 112..127

  float* cs = sa_so; float* ce = sa_se; float* cv = sa_vo; float* cw = sa_ve;
  float* ns = sb_so; float* ne = sb_se; float* nv = sb_vo; float* nw = sb_ve;

  for (int l = 0; l < 3; ++l) {
    // pack state (row-major A) + per-layer weights (fragment-blocked B)
    pack(cs, bf_so, NNODE, 32, NNODE, 32);
    pack(ce, bf_se, NNODE, 32, NNODE, 32);
    pack(cv, bf_vo, 3 * NNODE, 32, 3 * NNODE, 32);
    pack(cw, bf_ve, 3 * NNODE, 32, 3 * NNODE, 32);
    packB(lin1_so + l * 1024, wb_l1so, 32, 32, 32, 32);
    packB(lin1_se + l * 1024, wb_l1se, 32, 32, 32, 32);
    packB(lin1_vo + l * 1024, wb_l1vo, 32, 32, 32, 32);
    packB(lin1_ve + l * 1024, wb_l1ve, 32, 32, 32, 32);
    packB(sc_so + l * 1024, wb_wsso, 32, 32, 32, 32);
    packB(sc_se + l * 3072, wb_wsse, 32, 96, 32, 96);
    packB(sc_vo + l * 1024, wb_wsvo, 32, 32, 32, 32);
    packB(sc_ve + l * 1024, wb_wsve, 32, 32, 32, 32);
    packB(rad_w1 + l * 1000, wb_rw1, 10, 100, 32, 112);
    packB(rad_w2 + l * 32000, wb_rw2, 100, 320, 128, 320);
    packB(lin2_so + l * 2048, wb_w2so, 64, 32, 64, 32);
    packB(lin2_se + l * 6144, wb_w2se, 64, 96, 64, 96);
    packB(lin2_vo + l * 3072, wb_w2vo, 96, 32, 96, 32);
    packB(lin2_ve + l * 3072, wb_w2ve, 96, 32, 96, 32);

    // lin1 (WMMA)
    gemm(bf_so, wb_l1so, a_so, NNODE, 32, 32, 32, 32, ISQ32, 0);
    gemm(bf_se, wb_l1se, a_se, NNODE, 32, 32, 32, 32, ISQ32, 0);
    gemm(bf_vo, wb_l1vo, a_vo, 3 * NNODE, 32, 32, 32, 32, ISQ32, 0);
    gemm(bf_ve, wb_l1ve, a_ve, 3 * NNODE, 32, 32, 32, 32, ISQ32, 0);

    // radial MLP (WMMA, bf16 hidden)
    gemm_silu(emb_bf, wb_rw1, H_bf, NEDGE, 112, 32, 32, 128, ISQ10);
    gemm(H_bf, wb_rw2, Wrad, NEDGE, 320, 128, 128, 320, ISQ100, 0);

    // messages + scatter
    zero(n0o, accTotal);
    eq_msg<<<cdiv((long long)NEDGE * 32, 256), 256, 0, stream>>>(
        a_so, a_se, a_vo, a_ve, Wrad, y0, y1, esrc, edst, n0o, n0e, n1o, n1e);

    // aggregate features to bf16
    pack(n0o, bf_n0o, NNODE, 64, NNODE, 64);
    pack(n0e, bf_n0e, NNODE, 64, NNODE, 64);
    pack(n1o, bf_n1o, 3 * NNODE, 96, 3 * NNODE, 96);
    pack(n1e, bf_n1e, 3 * NNODE, 96, 3 * NNODE, 96);

    // output linears: self-connection + message path (accumulating WMMA)
    gemm(bf_so,  wb_wsso, o0o, NNODE, 32, 32, 32, 32, K_CS * ISQ32, 0);
    gemm(bf_n0o, wb_w2so, o0o, NNODE, 32, 64, 64, 32, K_CX * ISQ64, 1);
    gemm(bf_se,  wb_wsse, o0e, NNODE, 96, 32, 32, 96, K_CS * ISQ32, 0);
    gemm(bf_n0e, wb_w2se, o0e, NNODE, 96, 64, 64, 96, K_CX * ISQ64, 1);
    gemm(bf_vo,  wb_wsvo, o1o, 3 * NNODE, 32, 32, 32, 32, K_CS * ISQ32, 0);
    gemm(bf_n1o, wb_w2vo, o1o, 3 * NNODE, 32, 96, 96, 32, K_CX * ISQ96, 1);
    gemm(bf_ve,  wb_wsve, o1e, 3 * NNODE, 32, 32, 32, 32, K_CS * ISQ32, 0);
    gemm(bf_n1e, wb_w2ve, o1e, 3 * NNODE, 32, 96, 96, 32, K_CX * ISQ96, 1);

    eq_nonlin<<<cdiv((long long)NNODE * 32, 256), 256, 0, stream>>>(
        o0o, o0e, o1o, o1e, ns, ne, nv, nw);

    float* t;
    t = cs; cs = ns; ns = t;  t = ce; ce = ne; ne = t;
    t = cv; cv = nv; nv = t;  t = cw; cw = nw; nw = t;
  }

  // --- final readout ---
  pack(ce, bf_se, NNODE, 32, NNODE, 32);
  pack(cv, bf_vo, 3 * NNODE, 32, 3 * NNODE, 32);
  packB(flin1_se, wb_f1se, 32, 32, 32, 32);
  packB(flin1_vo, wb_f1vo, 32, 32, 32, 32);
  packB(frad_w1, wb_frw1, 10, 100, 32, 112);
  packB(frad_w2, wb_frw2, 100, 64, 128, 64);

  gemm(bf_se, wb_f1se, a_se, NNODE, 32, 32, 32, 32, ISQ32, 0);
  gemm(bf_vo, wb_f1vo, a_vo, 3 * NNODE, 32, 32, 32, 32, ISQ32, 0);
  gemm_silu(emb_bf, wb_frw1, H_bf, NEDGE, 112, 32, 32, 128, ISQ10);
  gemm(H_bf, wb_frw2, Wrad, NEDGE, 64, 128, 128, 64, ISQ100, 0);

  zero(n0e, (long long)NNODE * 64);
  eq_fmsg<<<cdiv((long long)NEDGE * 32, 256), 256, 0, stream>>>(
      a_se, a_vo, Wrad, y0, y1, esrc, edst, n0e);

  zero(out, 16);
  eq_readout<<<cdiv(NNODE, 256), 256, 0, stream>>>(ce, n0e, fsc_se, flin2, batch, out);
}
